// TransformerMoEBlock_6622839571051
// MI455X (gfx1250) — compile-verified, matrix operand order checked by reading, not run on hardware
//
#include <hip/hip_runtime.h>

// ---------------- types & constants ----------------
using u16 = unsigned short;
typedef __attribute__((ext_vector_type(16))) __bf16 v16bf;
typedef __attribute__((ext_vector_type(8)))  float  v8f;
typedef __attribute__((ext_vector_type(4)))  int    v4i;

constexpr int cB  = 2;
constexpr int cS  = 1024;
constexpr int cD  = 1024;
constexpr int cH  = 16;
constexpr int cHD = 64;
constexpr int cP  = 256;
constexpr int cE  = 8;
constexpr int cF  = 4096;        // 4*D
constexpr int cNS = cB * cS;     // 2048 tokens
constexpr int cN3 = 3 * cD;      // 3072
constexpr int CAP = 4352;        // MoE padded slot capacity (mult of 32, >= 4096+8*31)
constexpr int NT32 = CAP / 32;   // 136 row-tiles of 32

// ---------------- async global->LDS support (guarded) ----------------
#if defined(__has_builtin)
#if __has_builtin(__builtin_amdgcn_global_load_async_to_lds_b128) && \
    __has_builtin(__builtin_amdgcn_s_wait_asynccnt)
#define HAVE_ASYNC_LDS 1
#endif
#endif
#ifndef HAVE_ASYNC_LDS
#define HAVE_ASYNC_LDS 0
#endif

__device__ __forceinline__ void cp16_g2l(const u16* g, u16* l) {
#if HAVE_ASYNC_LDS
    // generic LDS address low 32 bits == LDS byte offset (ISA aperture rule);
    // generic global address == as(1) address.
    __builtin_amdgcn_global_load_async_to_lds_b128(
        (__attribute__((address_space(1))) v4i*)(unsigned long long)(uintptr_t)g,
        (__attribute__((address_space(3))) v4i*)(unsigned)(uintptr_t)l,
        0, 0);
#else
    *reinterpret_cast<uint4*>(l) = *reinterpret_cast<const uint4*>(g);
#endif
}
template <int N>
__device__ __forceinline__ void wait_async_le() {
#if HAVE_ASYNC_LDS
    __builtin_amdgcn_s_wait_asynccnt(N);
#endif
}

__device__ __forceinline__ u16 f2bf(float f) {
    unsigned u = __float_as_uint(f);
    unsigned r = (u + 0x7FFFu + ((u >> 16) & 1u)) >> 16;
    return (u16)r;
}
__device__ __forceinline__ float bf2f(u16 h) {
    return __uint_as_float(((unsigned)h) << 16);
}
__device__ __forceinline__ float geluf(float x) {
    return 0.5f * x * (1.f + tanhf(0.7978845608028654f * (x + 0.044715f * x * x * x)));
}
__device__ __forceinline__ float wred_sum(float v) {
    for (int o = 16; o; o >>= 1) v += __shfl_xor(v, o, 32);
    return v;
}
__device__ __forceinline__ float wred_max(float v) {
    for (int o = 16; o; o >>= 1) v = fmaxf(v, __shfl_xor(v, o, 32));
    return v;
}

// Fragment loader for 16-bit WMMA A (M x K row-major) / B supplied as Bt (N x K row-major).
// Lane L in [0,16): row = row0+L, K-runs at kk+0..7 and kk+16..23
// Lane L in [16,32): row = row0+L-16, K-runs at kk+8..15 and kk+24..31
__device__ __forceinline__ v16bf load_frag(const u16* base, int ld, int row0, int kk) {
    const int lane = threadIdx.x & 31;
    const u16* p = base + (size_t)(row0 + (lane & 15)) * (size_t)ld + kk + ((lane >> 4) << 3);
    union { v16bf v; uint4 q[2]; } u;
    u.q[0] = *reinterpret_cast<const uint4*>(p);
    u.q[1] = *reinterpret_cast<const uint4*>(p + 16);
    return u.v;
}

__device__ __forceinline__ v8f wmma_bf16(v16bf a, v16bf b, v8f c) {
    return __builtin_amdgcn_wmma_f32_16x16x32_bf16(false, a, false, b, (short)0, c, false, false);
}

// ---------------- tiny GEMM: ss = fp @ W + b  (B x P @ P x 2D) ----------------
__global__ void ss_gemm(const float* __restrict__ fp, const float* __restrict__ W,
                        const float* __restrict__ bias, float* __restrict__ out) {
    int idx = blockIdx.x * 256 + threadIdx.x;           // B*2048
    int b = idx >> 11, c = idx & 2047;
    float acc = bias[c];
    const float* f = fp + (size_t)b * cP;
    for (int p = 0; p < cP; ++p) acc += f[p] * W[(size_t)p * 2048 + c];
    out[idx] = acc;
}

// ---------------- transpose f32 [R x C] -> bf16 [C x R] ----------------
__global__ void transpose_to_bf16(const float* __restrict__ src, u16* __restrict__ dst,
                                  int R, int C, size_t sstride, size_t dstride) {
    const float* s = src + blockIdx.z * sstride;
    u16* d = dst + blockIdx.z * dstride;
    __shared__ float t[32][33];
    int c0 = blockIdx.x * 32, r0 = blockIdx.y * 32;
    int tx = threadIdx.x & 31, ty = threadIdx.x >> 5;    // 32 x 8
#pragma unroll
    for (int i = 0; i < 4; ++i) {
        int r = r0 + ty + 8 * i;
        t[ty + 8 * i][tx] = s[(size_t)r * C + c0 + tx];
    }
    __syncthreads();
#pragma unroll
    for (int i = 0; i < 4; ++i) {
        int c = c0 + ty + 8 * i;
        d[(size_t)c * R + r0 + tx] = f2bf(t[tx][ty + 8 * i]);
    }
}

// ---------------- adaLN: layernorm + (1+scale)*xn + shift ----------------
__global__ void adaln_apply(const float* __restrict__ xin, const float* __restrict__ ss,
                            float* __restrict__ outf, u16* __restrict__ outb) {
    int row = blockIdx.x;            // token (b*S+s)
    int b = row >> 10;
    const float* xr = xin + (size_t)row * cD;
    float v[4], s = 0.f, sq = 0.f;
#pragma unroll
    for (int i = 0; i < 4; ++i) {
        int c = threadIdx.x + 256 * i;
        float t = xr[c];
        v[i] = t; s += t; sq += t * t;
    }
    s = wred_sum(s); sq = wred_sum(sq);
    __shared__ float r0[8], r1[8];
    int w = threadIdx.x >> 5, lane = threadIdx.x & 31;
    if (lane == 0) { r0[w] = s; r1[w] = sq; }
    __syncthreads();
    float ts = 0.f, tq = 0.f;
#pragma unroll
    for (int j = 0; j < 8; ++j) { ts += r0[j]; tq += r1[j]; }
    float mu = ts * (1.f / 1024.f);
    float var = tq * (1.f / 1024.f) - mu * mu;
    float rn = rsqrtf(var + 1e-6f);
#pragma unroll
    for (int i = 0; i < 4; ++i) {
        int c = threadIdx.x + 256 * i;
        float sc = ss[(size_t)b * 2048 + c];
        float sh = ss[(size_t)b * 2048 + 1024 + c];
        float y = (v[i] - mu) * rn * (1.f + sc) + sh;
        if (outf) outf[(size_t)row * cD + c] = y;
        outb[(size_t)row * cD + c] = f2bf(y);
    }
}

// ---------------- generic bf16 WMMA GEMM with async-LDS double-buffered B ----------------
// C[M x N] = A[M x K] (bf16 row-major) * Bt[N x K]^T (bf16 N-major)
// Block: 256 threads = 8 waves arranged WM (rows) x (8/WM) (cols);
// each wave computes 16 rows x 16*NREP cols. B tile (BN x 32) staged in LDS.
// EPI: 0 = store f32, 1 = f32 + resid, 2 = store bf16, 3 = gelu -> bf16
template <int EPI, int WM, int NREP>
__global__ __launch_bounds__(256) void gemm_bf16(
    const u16* __restrict__ A, const u16* __restrict__ Bt,
    float* __restrict__ Cf, u16* __restrict__ Cb, const float* __restrict__ resid,
    int K, int lda, int ldb, int ldc,
    const int* __restrict__ tile_expert, size_t strideB) {
    constexpr int WN = 8 / WM;
    constexpr int BN = 16 * NREP * WN;    // block N-tile (rows of Bt staged per k-step)
    constexpr int SLD = 40;               // padded LDS row stride (elements) -> no bank conflicts
    constexpr int CH = BN / 64;           // 16B chunks copied per thread per tile
    __shared__ __align__(16) u16 sb[2][BN * SLD];

    const int w = threadIdx.x >> 5, lane = threadIdx.x & 31;
    const int wm = w % WM, wn = w / WM;
    const int m = blockIdx.y * (16 * WM) + wm * 16;
    const int nb0 = blockIdx.x * BN;
    const int nrel = wn * (16 * NREP);
    const u16* Bp = Bt + (tile_expert ? (size_t)tile_expert[blockIdx.y] * strideB : 0) +
                    (size_t)nb0 * ldb;

    auto stage = [&](int buf, int kk) {
#pragma unroll
        for (int c = 0; c < CH; ++c) {
            int id = threadIdx.x + 256 * c;
            int row = id >> 2, c16 = id & 3;
            cp16_g2l(Bp + (size_t)row * ldb + kk + c16 * 8,
                     &sb[buf][row * SLD + c16 * 8]);
        }
    };

    v8f zero = {};
    v8f acc[NREP];
#pragma unroll
    for (int j = 0; j < NREP; ++j) acc[j] = zero;

    stage(0, 0);
    for (int kk = 0; kk < K; kk += 32) {
        const int cur = (kk >> 5) & 1;
        const bool more = (kk + 32 < K);
        if (more) stage(cur ^ 1, kk + 32);
        if (more) wait_async_le<CH>(); else wait_async_le<0>();
        __syncthreads();
        v16bf a = load_frag(A, lda, m, kk);
        if (more)
            __builtin_prefetch(A + (size_t)(m + (lane & 15)) * lda + kk + 32, 0, 0);
#pragma unroll
        for (int j = 0; j < NREP; ++j) {
            v16bf b = load_frag(&sb[cur][0], SLD, nrel + 16 * j, 0);
            acc[j] = wmma_bf16(a, b, acc[j]);
        }
        __syncthreads();
    }
    const int rbase = 8 * (lane >> 4);
    const int col0 = lane & 15;
#pragma unroll
    for (int j = 0; j < NREP; ++j) {
#pragma unroll
        for (int r = 0; r < 8; ++r) {
            int row = m + r + rbase;
            size_t idx = (size_t)row * ldc + nb0 + nrel + 16 * j + col0;
            float v = acc[j][r];
            if constexpr (EPI == 0) Cf[idx] = v;
            else if constexpr (EPI == 1) Cf[idx] = v + resid[idx];
            else if constexpr (EPI == 2) Cb[idx] = f2bf(v);
            else Cb[idx] = f2bf(geluf(v));
        }
    }
}

// ---------------- RoPE + repack q,k,v ----------------
// qkv f32 [2048 x 3072] -> Qr,Kr bf16 [B,H,S,64]; Vt bf16 [B,H,64,S]
__global__ void rope_pack(const float* __restrict__ qkv, const float* __restrict__ freqs,
                          u16* __restrict__ Qr, u16* __restrict__ Kr, u16* __restrict__ Vt) {
    int idx = blockIdx.x * 256 + threadIdx.x;    // B*S*H*32
    int d2 = idx & 31;
    int h = (idx >> 5) & 15;
    int s = (idx >> 9) & 1023;
    int b = idx >> 19;
    size_t row = (size_t)(b * cS + s);
    const float* q = qkv + row * cN3 + h * cHD;
    const float* k = q + cD;
    const float* v = q + 2 * cD;
    float f = freqs[(size_t)s * 32 + d2];
    float c = cosf(f), sn = sinf(f);
    size_t bh = (size_t)(b * cH + h);
    size_t obase = (bh * cS + s) * cHD;
    float q1 = q[d2], q2 = q[d2 + 32];
    Qr[obase + d2]      = f2bf(q1 * c - q2 * sn);
    Qr[obase + d2 + 32] = f2bf(q1 * sn + q2 * c);
    float k1 = k[d2], k2 = k[d2 + 32];
    Kr[obase + d2]      = f2bf(k1 * c - k2 * sn);
    Kr[obase + d2 + 32] = f2bf(k1 * sn + k2 * c);
    size_t vb = bh * cHD * cS;
    Vt[vb + (size_t)d2 * cS + s]        = f2bf(v[d2]);
    Vt[vb + (size_t)(d2 + 32) * cS + s] = f2bf(v[d2 + 32]);
}

// ---------------- attention: 32 q-rows per workgroup, full softmax in LDS ----------------
constexpr int QT = 32;
constexpr int PLD = 1032;  // bf16 row stride for P (bank-conflict pad)
constexpr size_t ATTN_LDS = (size_t)QT * cS * 4 + (size_t)QT * PLD * 2;

__global__ __launch_bounds__(256) void attn_kernel(
    const u16* __restrict__ Qr, const u16* __restrict__ Kr, const u16* __restrict__ Vt,
    u16* __restrict__ attnO) {
    extern __shared__ char smem[];
    float* sc = (float*)smem;                                  // [QT][S]
    u16* pP = (u16*)(smem + (size_t)QT * cS * 4);              // [QT][PLD]

    const int qt = blockIdx.x, h = blockIdx.y, b = blockIdx.z;
    const size_t bh = (size_t)(b * cH + h);
    const int qbase = qt * QT;
    const u16* Qh = Qr + (bh * cS + qbase) * cHD;
    const u16* Kh = Kr + bh * cS * cHD;
    const u16* Vh = Vt + bh * cHD * cS;
    const int w = threadIdx.x >> 5, lane = threadIdx.x & 31;
    const int rbase = 8 * (lane >> 4), col0 = lane & 15;
    v8f zero = {};

    // Phase 1: scores = Q K^T * 0.125   (2 q-subtiles x 64 n-subtiles, 2 k-steps each)
    for (int t = w; t < 128; t += 8) {
        int qi = t & 1, nj = t >> 1;
        v8f acc = zero;
#pragma unroll
        for (int kk = 0; kk < cHD; kk += 32) {
            v16bf a = load_frag(Qh, cHD, qi * 16, kk);
            v16bf bfr = load_frag(Kh, cHD, nj * 16, kk);
            acc = wmma_bf16(a, bfr, acc);
        }
#pragma unroll
        for (int r = 0; r < 8; ++r)
            sc[(size_t)(qi * 16 + r + rbase) * cS + nj * 16 + col0] = acc[r] * 0.125f;
    }
    __syncthreads();

    // Phase 2: softmax per row -> bf16 P
    for (int rr = 0; rr < 4; ++rr) {
        int row = w * 4 + rr;
        float* srow = sc + (size_t)row * cS;
        float m = -3.4e38f;
        for (int j = 0; j < 32; ++j) m = fmaxf(m, srow[lane + 32 * j]);
        m = wred_max(m);
        float ssum = 0.f;
        for (int j = 0; j < 32; ++j) {
            float e = __expf(srow[lane + 32 * j] - m);
            srow[lane + 32 * j] = e;
            ssum += e;
        }
        ssum = wred_sum(ssum);
        float inv = 1.f / ssum;
        u16* prow = pP + (size_t)row * PLD;
        for (int j = 0; j < 32; ++j)
            prow[lane + 32 * j] = f2bf(srow[lane + 32 * j] * inv);
    }
    __syncthreads();

    // Phase 3: O = P V   (each wave one 16x16 tile of the 32x64 output)
    {
        int qi = w >> 2, dj = w & 3;
        v8f acc = zero;
        for (int kk = 0; kk < cS; kk += 32) {
            v16bf a = load_frag(pP, PLD, qi * 16, kk);
            v16bf bfr = load_frag(Vh, cS, dj * 16, kk);
            acc = wmma_bf16(a, bfr, acc);
        }
#pragma unroll
        for (int r = 0; r < 8; ++r) {
            int s = qbase + qi * 16 + r + rbase;
            int col = h * cHD + dj * 16 + col0;
            attnO[(size_t)(b * cS + s) * cD + col] = f2bf(acc[r]);
        }
    }
}

// ---------------- router: logits, top-2, gates ----------------
__global__ void router_kernel(const float* __restrict__ xln2f, const float* __restrict__ wr,
                              const float* __restrict__ rbias, float* __restrict__ gates,
                              int* __restrict__ topk, int* __restrict__ choice,
                              unsigned* __restrict__ counts) {
    int n = blockIdx.x * 256 + threadIdx.x;   // 2048 tokens
    const float* xr = xln2f + (size_t)n * cD;
    float lg[cE];
#pragma unroll
    for (int e = 0; e < cE; ++e) lg[e] = 0.f;
    for (int d = 0; d < cD; ++d) {
        float xv = xr[d];
        const float* wrow = wr + (size_t)d * cE;
#pragma unroll
        for (int e = 0; e < cE; ++e) lg[e] += xv * wrow[e];
    }
    float scv[cE];
#pragma unroll
    for (int e = 0; e < cE; ++e) scv[e] = lg[e] + rbias[e];
    int i0 = 0;
#pragma unroll
    for (int e = 1; e < cE; ++e) if (scv[e] > scv[i0]) i0 = e;
    int i1 = (i0 == 0) ? 1 : 0;
#pragma unroll
    for (int e = 0; e < cE; ++e)
        if (e != i0 && scv[e] > scv[i1]) i1 = e;
    float l0 = lg[i0], l1 = lg[i1];
    float m = fmaxf(l0, l1);
    float e0 = __expf(l0 - m), e1 = __expf(l1 - m);
    float inv = 1.f / (e0 + e1);
    gates[n * 2] = e0 * inv;  gates[n * 2 + 1] = e1 * inv;
    topk[n * 2] = i0;         topk[n * 2 + 1] = i1;
    choice[n * 2] = i0;       choice[n * 2 + 1] = i1;
    atomicAdd(&counts[i0], 1u);
    atomicAdd(&counts[i1], 1u);
}

// ---------------- scatter machinery ----------------
__global__ void moe_init(unsigned* counts, unsigned* cursor, int* token_of) {
    int idx = blockIdx.x * 256 + threadIdx.x;
    if (idx < cE) { counts[idx] = 0u; cursor[idx] = 0u; }
    if (idx < CAP) token_of[idx] = -1;
}
__global__ void moe_offsets(const unsigned* counts, unsigned* offsets, int* tile_expert) {
    unsigned off = 0;
    for (int t = 0; t < NT32; ++t) tile_expert[t] = 0;
    for (int e = 0; e < cE; ++e) {
        offsets[e] = off;
        unsigned pad = (counts[e] + 31u) & ~31u;
        for (unsigned t = off / 32u; t < (off + pad) / 32u; ++t) tile_expert[t] = e;
        off += pad;
    }
}
__global__ void moe_assign(const int* choice, const unsigned* offsets, unsigned* cursor,
                           int* slot_of, int* token_of) {
    int idx = blockIdx.x * 256 + threadIdx.x;   // 4096
    int e = choice[idx];
    unsigned r = atomicAdd(&cursor[e], 1u);
    int slot = (int)(offsets[e] + r);
    slot_of[idx] = slot;
    token_of[slot] = idx >> 1;
}
__global__ void moe_gather(const int* __restrict__ token_of, const u16* __restrict__ xln2b,
                           u16* __restrict__ Xg) {
    int slot = blockIdx.x;
    int n = token_of[slot];
#pragma unroll
    for (int i = 0; i < 4; ++i) {
        int c = threadIdx.x + 256 * i;
        Xg[(size_t)slot * cD + c] = (n >= 0) ? xln2b[(size_t)n * cD + c] : (u16)0;
    }
}
__global__ void moe_combine(const float* __restrict__ x1, const float* __restrict__ Yg,
                            const int* __restrict__ slot_of, const float* __restrict__ gates,
                            float* __restrict__ out) {
    int idx = blockIdx.x * 256 + threadIdx.x;   // 2048*1024
    int n = idx >> 10, c = idx & 1023;
    int s0 = slot_of[n * 2], s1 = slot_of[n * 2 + 1];
    float g0 = gates[n * 2], g1 = gates[n * 2 + 1];
    out[idx] = x1[idx] + g0 * Yg[(size_t)s0 * cD + c] + g1 * Yg[(size_t)s1 * cD + c];
}

// ---------------- host launch ----------------
extern "C" void kernel_launch(void* const* d_in, const int* in_sizes, int n_in,
                              void* d_out, int out_size, void* d_ws, size_t ws_size,
                              hipStream_t stream) {
    const float* x        = (const float*)d_in[0];
    const float* freqs    = (const float*)d_in[1];
    const float* fp       = (const float*)d_in[2];
    const float* aln1_w   = (const float*)d_in[3];
    const float* aln1_b   = (const float*)d_in[4];
    const float* aln2_w   = (const float*)d_in[5];
    const float* aln2_b   = (const float*)d_in[6];
    const float* wqkv     = (const float*)d_in[7];
    const float* wo       = (const float*)d_in[8];
    const float* w_router = (const float*)d_in[9];
    const float* rbias    = (const float*)d_in[10];
    const float* w1       = (const float*)d_in[11];
    const float* w2       = (const float*)d_in[12];

    float* out_x     = (float*)d_out;
    float* out_gates = out_x + (size_t)cNS * cD;            // 2,097,152
    int*   out_topk  = (int*)(out_gates + (size_t)cNS * 2); // +4096

    char* w = (char*)d_ws;
    size_t off = 0;
    auto alloc = [&](size_t bytes) {
        size_t o = off;
        off = (off + bytes + 255) & ~(size_t)255;
        return o;
    };
    float* ss1    = (float*)(w + alloc((size_t)cB * 2048 * 4));
    float* ss2    = (float*)(w + alloc((size_t)cB * 2048 * 4));
    u16*   xln1b  = (u16*)  (w + alloc((size_t)cNS * cD * 2));
    u16*   wqkvT  = (u16*)  (w + alloc((size_t)cN3 * cD * 2));
    u16*   woT    = (u16*)  (w + alloc((size_t)cD * cD * 2));
    u16*   w1T    = (u16*)  (w + alloc((size_t)cE * cF * cD * 2));
    u16*   w2T    = (u16*)  (w + alloc((size_t)cE * cD * cF * 2));
    float* qkvf   = (float*)(w + alloc((size_t)cNS * cN3 * 4));   // reused as Yg later
    u16*   Qr     = (u16*)  (w + alloc((size_t)cB * cH * cS * cHD * 2));
    u16*   Kr     = (u16*)  (w + alloc((size_t)cB * cH * cS * cHD * 2));
    u16*   Vt     = (u16*)  (w + alloc((size_t)cB * cH * cHD * cS * 2));
    u16*   attnOb = (u16*)  (w + alloc((size_t)cNS * cD * 2));
    float* x1     = (float*)(w + alloc((size_t)cNS * cD * 4));
    float* xln2f  = (float*)(w + alloc((size_t)cNS * cD * 4));
    u16*   xln2b  = (u16*)  (w + alloc((size_t)cNS * cD * 2));
    u16*   Xg     = (u16*)  (w + alloc((size_t)CAP * cD * 2));
    u16*   Hb     = (u16*)  (w + alloc((size_t)CAP * cF * 2));
    unsigned* counts  = (unsigned*)(w + alloc(cE * 4));
    unsigned* cursor  = (unsigned*)(w + alloc(cE * 4));
    unsigned* offsets = (unsigned*)(w + alloc(cE * 4));
    int* tile_expert  = (int*)(w + alloc(NT32 * 4));
    int* token_of     = (int*)(w + alloc(CAP * 4));
    int* slot_of      = (int*)(w + alloc(cNS * 2 * 4));
    int* choice       = (int*)(w + alloc(cNS * 2 * 4));
    float* Yg = qkvf;  // alias: qkv buffer dead after rope_pack

    // 1) adaLN modulation vectors
    ss_gemm<<<dim3(cB * 2048 / 256), 256, 0, stream>>>(fp, aln1_w, aln1_b, ss1);
    ss_gemm<<<dim3(cB * 2048 / 256), 256, 0, stream>>>(fp, aln2_w, aln2_b, ss2);

    // 2) weight transposes to bf16 N-major
    transpose_to_bf16<<<dim3(cN3 / 32, cD / 32, 1), 256, 0, stream>>>(wqkv, wqkvT, cD, cN3, 0, 0);
    transpose_to_bf16<<<dim3(cD / 32, cD / 32, 1), 256, 0, stream>>>(wo, woT, cD, cD, 0, 0);
    transpose_to_bf16<<<dim3(cF / 32, cD / 32, cE), 256, 0, stream>>>(
        w1, w1T, cD, cF, (size_t)cD * cF, (size_t)cF * cD);
    transpose_to_bf16<<<dim3(cD / 32, cF / 32, cE), 256, 0, stream>>>(
        w2, w2T, cF, cD, (size_t)cF * cD, (size_t)cD * cF);

    // 3) adaLN1 -> bf16
    adaln_apply<<<dim3(cNS), 256, 0, stream>>>(x, ss1, nullptr, xln1b);

    // 4) QKV = xln1 @ wqkv
    gemm_bf16<0, 8, 4><<<dim3(cN3 / 64, cNS / 128), 256, 0, stream>>>(
        xln1b, wqkvT, qkvf, nullptr, nullptr, cD, cD, cD, cN3, nullptr, 0);

    // 5) RoPE + pack
    rope_pack<<<dim3(cB * cS * cH * 32 / 256), 256, 0, stream>>>(qkvf, freqs, Qr, Kr, Vt);

    // 6) attention
    attn_kernel<<<dim3(cS / QT, cH, cB), 256, ATTN_LDS, stream>>>(Qr, Kr, Vt, attnOb);

    // 7) x1 = attnO @ wo + x
    gemm_bf16<1, 8, 4><<<dim3(cD / 64, cNS / 128), 256, 0, stream>>>(
        attnOb, woT, x1, nullptr, x, cD, cD, cD, cD, nullptr, 0);

    // 8) adaLN2 -> f32 + bf16
    adaln_apply<<<dim3(cNS), 256, 0, stream>>>(x1, ss2, xln2f, xln2b);

    // 9) MoE routing + scatter
    moe_init<<<dim3((CAP + 255) / 256), 256, 0, stream>>>(counts, cursor, token_of);
    router_kernel<<<dim3(cNS / 256), 256, 0, stream>>>(
        xln2f, w_router, rbias, out_gates, out_topk, choice, counts);
    moe_offsets<<<dim3(1), 1, 0, stream>>>(counts, offsets, tile_expert);
    moe_assign<<<dim3(cNS * 2 / 256), 256, 0, stream>>>(choice, offsets, cursor, slot_of, token_of);
    moe_gather<<<dim3(CAP), 256, 0, stream>>>(token_of, xln2b, Xg);

    // 10) FFN1: H = gelu(Xg @ w1[e]) -> bf16
    gemm_bf16<3, 2, 4><<<dim3(cF / 256, NT32), 256, 0, stream>>>(
        Xg, w1T, nullptr, Hb, nullptr, cD, cD, cD, cF, tile_expert, (size_t)cF * cD);

    // 11) FFN2: Yg = H @ w2[e] -> f32
    gemm_bf16<0, 2, 4><<<dim3(cD / 256, NT32), 256, 0, stream>>>(
        Hb, w2T, Yg, nullptr, nullptr, cF, cF, cF, cD, tile_expert, (size_t)cD * cF);

    // 12) out = x1 + g0*Y[slot0] + g1*Y[slot1]
    moe_combine<<<dim3((size_t)cNS * cD / 256), 256, 0, stream>>>(
        x1, Yg, slot_of, out_gates, out_x);

    (void)in_sizes; (void)n_in; (void)out_size; (void)ws_size;
}
